// AttnBlock_67980742361809
// MI455X (gfx1250) — matrix operand order repaired and checked by reading, not compiled, not run on hardware
//
#include <hip/hip_runtime.h>
#include <hip/hip_bf16.h>
#include <math.h>

typedef __attribute__((ext_vector_type(16))) _Float16 v16h;
typedef __attribute__((ext_vector_type(8)))  float    v8f;
typedef __attribute__((ext_vector_type(8)))  _Float16 h8;
typedef __attribute__((ext_vector_type(4)))  _Float16 h4;

#define BB     4
#define IMG    112
#define NTOK   (IMG*IMG)        // 12544
#define BN     (BB*NTOK)        // 50176
#define CDIM   256
#define NHEAD  4                // GH == LH == 4
#define HD     32
#define GD     128              // = LD
#define HID    1024
#define NPOOL  256              // pooled tokens per batch
#define NWIN   256              // 7x7 windows per batch
#define SCALE  0.17677669529663687f
#define WSZ    7

// ---------------------------------------------------------------------------
// Fragment loaders (guaranteed b128 vector loads).
// A-frag (16x32, contiguous k order, not ISA table order): per-lane elements are memory [0..15]
// relative to (row base + half*16). B-frag contiguous: 16 consecutive halfs.
// ---------------------------------------------------------------------------
__device__ __forceinline__ v16h ld_frag_a(const _Float16* p) {
    h8 lo = *(const h8*)p;
    h8 hi = *(const h8*)(p + 8);
    return __builtin_shufflevector(lo, hi, 0,1,2,3,4,5,6,7,8,9,10,11,12,13,14,15);
}
__device__ __forceinline__ v16h ld_frag_c16(const _Float16* p) {
    h8 lo = *(const h8*)p;
    h8 hi = *(const h8*)(p + 8);
    return __builtin_shufflevector(lo, hi, 0,1,2,3,4,5,6,7,8,9,10,11,12,13,14,15);
}
#define WMMA(a,b,c) __builtin_amdgcn_wmma_f32_16x16x32_f16(false, (a), false, (b), (short)0, (c), false, false)

// ---------------------------------------------------------------------------
// fp32 -> f16 weight conversion with fragment swizzle:
// dst[((kb*(N/16)+nb)*2 + half)*256 + n*16 + e] = src[kb*32+half*16+e][nb*16+n]
// ---------------------------------------------------------------------------
__global__ void cvt_swz_kernel(const float* __restrict__ src, _Float16* __restrict__ dst,
                               int K, int N) {
    int i = blockIdx.x * 256 + threadIdx.x;
    if (i >= K * N) return;
    int k = i / N, c = i - k * N;
    int kb = k >> 5, half = (k >> 4) & 1, e = k & 15;
    int nb = c >> 4, n = c & 15;
    dst[(((size_t)kb * (N >> 4) + nb) * 2 + half) * 256 + n * 16 + e] = (_Float16)src[i];
}

// ---------------------------------------------------------------------------
// LayerNorm over 256 channels -> f16. One wave per row.
// ---------------------------------------------------------------------------
__global__ void __launch_bounds__(256) ln_f16_kernel(const float* __restrict__ X,
                                                     const float* __restrict__ g,
                                                     const float* __restrict__ b,
                                                     _Float16* __restrict__ Y) {
    int wave = threadIdx.x >> 5, lane = threadIdx.x & 31;
    int row  = blockIdx.x * 8 + wave;
    const float* xr = X + (size_t)row * CDIM + lane * 8;
    float v[8], s = 0.f, s2 = 0.f;
#pragma unroll
    for (int i = 0; i < 8; ++i) { v[i] = xr[i]; s += v[i]; s2 += v[i] * v[i]; }
#pragma unroll
    for (int off = 16; off >= 1; off >>= 1) { s += __shfl_xor(s, off, 32); s2 += __shfl_xor(s2, off, 32); }
    float mean = s * (1.f / CDIM);
    float rstd = rsqrtf(s2 * (1.f / CDIM) - mean * mean + 1e-5f);
    h8 o;
#pragma unroll
    for (int i = 0; i < 8; ++i)
        o[i] = (_Float16)((v[i] - mean) * rstd * g[lane * 8 + i] + b[lane * 8 + i]);
    *(h8*)(Y + (size_t)row * CDIM + lane * 8) = o;
}

// ---------------------------------------------------------------------------
// 7x7 average pool of xrn (f16 -> f16). One block per pooled token.
// ---------------------------------------------------------------------------
__global__ void __launch_bounds__(256) pool_kernel(const _Float16* __restrict__ xrn,
                                                   _Float16* __restrict__ pooled) {
    int tok = blockIdx.x;
    int b = tok >> 8, t = tok & 255;
    int gy = t >> 4, gx = t & 15;
    int c = threadIdx.x;
    float s = 0.f;
#pragma unroll
    for (int iy = 0; iy < WSZ; ++iy)
        for (int ix = 0; ix < WSZ; ++ix) {
            int n = (gy * WSZ + iy) * IMG + gx * WSZ + ix;
            s += (float)xrn[((size_t)b * NTOK + n) * CDIM + c];
        }
    pooled[(size_t)tok * CDIM + c] = (_Float16)(s * (1.f / 49.f));
}

// ---------------------------------------------------------------------------
// WMMA GEMM, no LDS: C[M,N] = A[M,K](f16) * Wsw(K,N swizzled f16) + bias.
// Block 256 thr = 8 waves -> 256 rows x 64 cols; wave = 32 rows x 64 cols.
// Weights come pre-swizzled so every B-fragment = 2 contiguous b128 loads.
// Epilogue: f16 store, or fp32 residual store into 256-wide buffer.
// M%256==0, N%64==0, K%32==0 for every call here.
// ---------------------------------------------------------------------------
__global__ void __launch_bounds__(256) gemm_wmma_kernel(const _Float16* __restrict__ A,
                                                        const _Float16* __restrict__ Wsw,
                                                        const float* __restrict__ bias,
                                                        _Float16* __restrict__ OutH,
                                                        float* __restrict__ OutF,
                                                        const float* __restrict__ Res,
                                                        int N, int K, int colOff) {
    int wave = threadIdx.x >> 5, lane = threadIdx.x & 31;
    int m = lane & 15, half = lane >> 4;
    int row0 = blockIdx.x * 256 + wave * 32;
    int col0 = blockIdx.y * 64;
    v8f acc[2][4] = {};
    const _Float16* a0 = A + (size_t)(row0 + m) * K + half * 16;
    const _Float16* a1 = a0 + (size_t)16 * K;
    int NB = N >> 4;
    const _Float16* wb = Wsw + (size_t)(col0 >> 4) * 512 + half * 256 + m * 16;

    for (int kb = 0; kb < (K >> 5); ++kb) {
        v16h aA = ld_frag_a(a0 + kb * 32);
        v16h aB = ld_frag_a(a1 + kb * 32);
        __builtin_prefetch(a0 + kb * 32 + 64, 0, 1);
        const _Float16* wk = wb + (size_t)kb * NB * 512;
#pragma unroll
        for (int nt = 0; nt < 4; ++nt) {
            v16h bf = ld_frag_c16(wk + nt * 512);
            acc[0][nt] = WMMA(aA, bf, acc[0][nt]);
            acc[1][nt] = WMMA(aB, bf, acc[1][nt]);
        }
    }
#pragma unroll
    for (int rr = 0; rr < 2; ++rr)
#pragma unroll
        for (int nt = 0; nt < 4; ++nt)
#pragma unroll
            for (int g = 0; g < 8; ++g) {
                int r = row0 + rr * 16 + half * 8 + g;
                int c = col0 + nt * 16 + m;
                float v = acc[rr][nt][g] + bias[c];
                if (OutF) {
                    size_t o = (size_t)r * CDIM + colOff + c;
                    OutF[o] = Res[o] + v;
                } else {
                    OutH[(size_t)r * (size_t)N + c] = (_Float16)v;
                }
            }
}

// ---------------------------------------------------------------------------
// Transpose global-attention V: vT[((b*4+h)*32+d)*256 + key] = kv[b,key,128+h*32+d]
// ---------------------------------------------------------------------------
__global__ void vtrans_kernel(const _Float16* __restrict__ kvb, _Float16* __restrict__ vT) {
    int i = blockIdx.x * 256 + threadIdx.x;   // 4*4*32*256 = 131072
    int key = i & 255, r = i >> 8;            // r = (b*4+h)*32+d
    int b = r >> 7, hd = r & 127;
    vT[(size_t)r * 256 + key] = kvb[((size_t)b * NPOOL + key) * CDIM + GD + hd];
}

// ---------------------------------------------------------------------------
// Global attention. One wave per (b, head, 16-query tile); 4 waves/block.
// ---------------------------------------------------------------------------
__global__ void __launch_bounds__(128) gattn_kernel(const _Float16* __restrict__ q,
                                                    const _Float16* __restrict__ kv,
                                                    const _Float16* __restrict__ vT,
                                                    _Float16* __restrict__ gout) {
    __shared__ _Float16 P[4][16][256];
    __shared__ float rmaxL[4][16];
    __shared__ float rsumL[4][32];
    int wave = threadIdx.x >> 5, lane = threadIdx.x & 31;
    int m = lane & 15, half = lane >> 4;
    int b = blockIdx.z, h = blockIdx.y;
    int row0 = (blockIdx.x * 4 + wave) * 16;
    const _Float16* qp  = q  + ((size_t)b * NTOK + row0) * GD + h * HD;
    const _Float16* kp  = kv + (size_t)b * NPOOL * CDIM + h * HD;
    const _Float16* vTp = vT + ((size_t)(b * NHEAD + h) * HD) * 256;

    v16h aq = ld_frag_a(qp + (size_t)m * GD + half * 16);
    float rmax[8];
#pragma unroll
    for (int g = 0; g < 8; ++g) rmax[g] = -1e30f;

    for (int kt = 0; kt < 16; ++kt) {
        v16h bk = ld_frag_c16(kp + (size_t)(kt * 16 + m) * CDIM + half * 16);
        v8f s = {};
        s = WMMA(aq, bk, s);
#pragma unroll
        for (int g = 0; g < 8; ++g) {
            float v = s[g] * SCALE;
            rmax[g] = fmaxf(rmax[g], v);
            P[wave][half * 8 + g][kt * 16 + m] = (_Float16)v;
        }
    }
#pragma unroll
    for (int off = 1; off <= 8; off <<= 1)
#pragma unroll
        for (int g = 0; g < 8; ++g) rmax[g] = fmaxf(rmax[g], __shfl_xor(rmax[g], off, 32));
    if (m == 0)
#pragma unroll
        for (int g = 0; g < 8; ++g) rmaxL[wave][half * 8 + g] = rmax[g];
    __syncthreads();

    // exp pass: lane owns row m, half the keys; vectorized 8-wide
    {
        float rm = rmaxL[wave][m];
        float ps = 0.f;
        _Float16* prow = &P[wave][m][half * 128];
#pragma unroll 4
        for (int kk = 0; kk < 128; kk += 8) {
            h8 v = *(h8*)(prow + kk);
            h8 o;
#pragma unroll
            for (int j = 0; j < 8; ++j) {
                float f = __expf((float)v[j] - rm);
                ps += f;
                o[j] = (_Float16)f;
            }
            *(h8*)(prow + kk) = o;
        }
        rsumL[wave][half * 16 + m] = ps;
    }
    __syncthreads();

    v8f o[2] = {};
    for (int kt2 = 0; kt2 < 8; ++kt2) {
        v16h apv = ld_frag_a(&P[wave][m][kt2 * 32 + half * 16]);
#pragma unroll
        for (int nt = 0; nt < 2; ++nt) {
            v16h bv = ld_frag_c16(vTp + (size_t)(nt * 16 + m) * 256 + kt2 * 32 + half * 16);
            o[nt] = WMMA(apv, bv, o[nt]);
        }
    }
#pragma unroll
    for (int g = 0; g < 8; ++g) {
        int rr = half * 8 + g;
        float inv = 1.f / (rsumL[wave][rr] + rsumL[wave][16 + rr]);
#pragma unroll
        for (int nt = 0; nt < 2; ++nt)
            gout[((size_t)b * NTOK + row0 + rr) * GD + h * HD + nt * 16 + m] =
                (_Float16)(o[nt][g] * inv);
    }
}

// ---------------------------------------------------------------------------
// Window repack: gather ql/kvl rows into padded window-major buffers.
// qlw/klw: [b*256+w][64][128] (t>=49 zero).  vlwT: [b*256+w][128 ch][64 t].
// ---------------------------------------------------------------------------
__device__ __forceinline__ int win_row(int gy, int gx, int t) {
    return (gy * WSZ + t / WSZ) * IMG + gx * WSZ + t % WSZ;
}

__global__ void __launch_bounds__(256) lrepack_kernel(const _Float16* __restrict__ ql,
                                                      const _Float16* __restrict__ kvl,
                                                      _Float16* __restrict__ qlw,
                                                      _Float16* __restrict__ klw,
                                                      _Float16* __restrict__ vlwT) {
    int w = blockIdx.x;                 // 0..1023
    int b = w >> 8, g0 = w & 255;
    int gy = g0 >> 4, gx = g0 & 15;
    int t = threadIdx.x >> 2;           // 0..63
    int cseg = (threadIdx.x & 3) * 32;  // 0,32,64,96
    bool valid = t < 49;
    size_t srow = valid ? ((size_t)b * NTOK + win_row(gy, gx, t)) : 0;
    _Float16* qdst = qlw + ((size_t)w * 64 + t) * GD + cseg;
    _Float16* kdst = klw + ((size_t)w * 64 + t) * GD + cseg;
#pragma unroll
    for (int j0 = 0; j0 < 32; j0 += 8) {
        h8 qv = {}, kv8 = {};
        if (valid) {
            qv  = *(const h8*)(ql  + srow * GD   + cseg + j0);
            kv8 = *(const h8*)(kvl + srow * CDIM + cseg + j0);
        }
        *(h8*)(qdst + j0) = qv;
        *(h8*)(kdst + j0) = kv8;
#pragma unroll
        for (int j = 0; j < 8; ++j) {
            _Float16 vv = valid ? kvl[srow * CDIM + GD + cseg + j0 + j] : (_Float16)0;
            vlwT[((size_t)w * GD + cseg + j0 + j) * 64 + t] = vv;
        }
    }
}

// ---------------------------------------------------------------------------
// Local window attention (49 tokens padded to 64) on repacked buffers.
// One wave per (b, window, head); 4 waves/block.
// ---------------------------------------------------------------------------
__global__ void __launch_bounds__(128) lattn_kernel(const _Float16* __restrict__ qlw,
                                                    const _Float16* __restrict__ klw,
                                                    const _Float16* __restrict__ vlwT,
                                                    _Float16* __restrict__ lout) {
    __shared__ _Float16 P[4][16][64];
    int wave = threadIdx.x >> 5, lane = threadIdx.x & 31;
    int m = lane & 15, half = lane >> 4;
    int b = blockIdx.z, h = blockIdx.y;
    int g0 = blockIdx.x * 4 + wave;
    int w = b * NWIN + g0;
    int gy = g0 >> 4, gx = g0 & 15;
    const _Float16* qbase = qlw  + (size_t)w * 64 * GD + h * HD;
    const _Float16* kbase = klw  + (size_t)w * 64 * GD + h * HD;
    const _Float16* vbase = vlwT + (size_t)w * GD * 64 + (size_t)h * HD * 64;

    for (int qt = 0; qt < 4; ++qt) {
        v16h aq = ld_frag_a(qbase + (size_t)(qt * 16 + m) * GD + half * 16);
        float p[4][8];
        float rmax[8];
#pragma unroll
        for (int g = 0; g < 8; ++g) rmax[g] = -1e30f;
#pragma unroll
        for (int kt = 0; kt < 4; ++kt) {
            v16h bk = ld_frag_c16(kbase + (size_t)(kt * 16 + m) * GD + half * 16);
            v8f s = {};
            s = WMMA(aq, bk, s);
            bool oob = (kt * 16 + m) >= 49;
#pragma unroll
            for (int g = 0; g < 8; ++g) {
                float v = oob ? -1e30f : s[g] * SCALE;
                p[kt][g] = v;
                rmax[g] = fmaxf(rmax[g], v);
            }
        }
#pragma unroll
        for (int off = 1; off <= 8; off <<= 1)
#pragma unroll
            for (int g = 0; g < 8; ++g) rmax[g] = fmaxf(rmax[g], __shfl_xor(rmax[g], off, 32));
        float rsum[8];
#pragma unroll
        for (int g = 0; g < 8; ++g) rsum[g] = 0.f;
#pragma unroll
        for (int kt = 0; kt < 4; ++kt)
#pragma unroll
            for (int g = 0; g < 8; ++g) {
                float v = __expf(p[kt][g] - rmax[g]);
                rsum[g] += v;
                P[wave][half * 8 + g][kt * 16 + m] = (_Float16)v;
            }
#pragma unroll
        for (int off = 1; off <= 8; off <<= 1)
#pragma unroll
            for (int g = 0; g < 8; ++g) rsum[g] += __shfl_xor(rsum[g], off, 32);
        __syncthreads();

        v8f o[2] = {};
#pragma unroll
        for (int kt2 = 0; kt2 < 2; ++kt2) {
            v16h apv = ld_frag_a(&P[wave][m][kt2 * 32 + half * 16]);
#pragma unroll
            for (int nt = 0; nt < 2; ++nt) {
                v16h bv = ld_frag_c16(vbase + (size_t)(nt * 16 + m) * 64 + kt2 * 32 + half * 16);
                o[nt] = WMMA(apv, bv, o[nt]);
            }
        }
#pragma unroll
        for (int g = 0; g < 8; ++g) {
            int tq = qt * 16 + half * 8 + g;
            if (tq < 49) {
                int n = win_row(gy, gx, tq);
                float inv = 1.f / rsum[g];
#pragma unroll
                for (int nt = 0; nt < 2; ++nt)
                    lout[((size_t)b * NTOK + n) * GD + h * HD + nt * 16 + m] =
                        (_Float16)(o[nt][g] * inv);
            }
        }
        __syncthreads();
    }
}

// ---------------------------------------------------------------------------
// Depthwise 3x3 SAME conv + bias + exact GELU, channels-last.
// ---------------------------------------------------------------------------
__global__ void __launch_bounds__(256) dwconv_gelu_kernel(const _Float16* __restrict__ Hin,
                                                          const float* __restrict__ Wd,
                                                          const float* __restrict__ Bd,
                                                          _Float16* __restrict__ Hout) {
    int x = blockIdx.x, y = blockIdx.y, b = blockIdx.z;
    int c = threadIdx.x * 4;
    float acc[4] = {0.f, 0.f, 0.f, 0.f};
#pragma unroll
    for (int dy = -1; dy <= 1; ++dy) {
        int yy = y + dy;
        if (yy < 0 || yy >= IMG) continue;
#pragma unroll
        for (int dx = -1; dx <= 1; ++dx) {
            int xx = x + dx;
            if (xx < 0 || xx >= IMG) continue;
            h4 v = *(const h4*)(Hin + (((size_t)b * IMG + yy) * IMG + xx) * HID + c);
            int widx = (dy + 1) * 3 + (dx + 1);
#pragma unroll
            for (int j = 0; j < 4; ++j)
                acc[j] += (float)v[j] * Wd[(c + j) * 9 + widx];
        }
    }
    h4 outv;
#pragma unroll
    for (int j = 0; j < 4; ++j) {
        float t = acc[j] + Bd[c + j];
        outv[j] = (_Float16)(0.5f * t * (1.f + erff(t * 0.70710678118f)));
    }
    *(h4*)(Hout + (((size_t)b * IMG + y) * IMG + x) * HID + c) = outv;
}

// ---------------------------------------------------------------------------
// Host-side launch
// ---------------------------------------------------------------------------
extern "C" void kernel_launch(void* const* d_in, const int* in_sizes, int n_in,
                              void* d_out, int out_size, void* d_ws, size_t ws_size,
                              hipStream_t stream) {
    const float* x    = (const float*)d_in[0];
    const float* xrev = (const float*)d_in[1];
    const float* g1   = (const float*)d_in[2];
    const float* b1   = (const float*)d_in[3];
    const float* g2   = (const float*)d_in[4];
    const float* b2   = (const float*)d_in[5];
    const float* GqW  = (const float*)d_in[6];   const float* Gqb  = (const float*)d_in[7];
    const float* GkvW = (const float*)d_in[8];   const float* Gkvb = (const float*)d_in[9];
    const float* GpW  = (const float*)d_in[10];  const float* Gpb  = (const float*)d_in[11];
    const float* LqW  = (const float*)d_in[12];  const float* Lqb  = (const float*)d_in[13];
    const float* LkvW = (const float*)d_in[14];  const float* Lkvb = (const float*)d_in[15];
    const float* LpW  = (const float*)d_in[16];  const float* Lpb  = (const float*)d_in[17];
    const float* f1W  = (const float*)d_in[18];  const float* f1b  = (const float*)d_in[19];
    const float* dwW  = (const float*)d_in[20];  const float* dwb  = (const float*)d_in[21];
    const float* f2W  = (const float*)d_in[22];  const float* f2b  = (const float*)d_in[23];
    float* out = (float*)d_out;

    char* ws = (char*)d_ws;
    size_t off = 0;
    auto alloc = [&](size_t bytes) { void* p = ws + off; off += (bytes + 255) & ~(size_t)255; return p; };
    _Float16* xn     = (_Float16*)alloc((size_t)BN * CDIM * 2);
    _Float16* xrn    = (_Float16*)alloc((size_t)BN * CDIM * 2);
    _Float16* pooled = (_Float16*)alloc((size_t)BB * NPOOL * CDIM * 2);
    _Float16* qb     = (_Float16*)alloc((size_t)BN * GD * 2);
    _Float16* kvb    = (_Float16*)alloc((size_t)BB * NPOOL * CDIM * 2);
    _Float16* vT     = (_Float16*)alloc((size_t)BB * NHEAD * HD * 256 * 2);
    _Float16* qlb    = (_Float16*)alloc((size_t)BN * GD * 2);
    _Float16* kvlb   = (_Float16*)alloc((size_t)BN * CDIM * 2);
    _Float16* qlw    = (_Float16*)alloc((size_t)BB * NWIN * 64 * GD * 2);
    _Float16* klw    = (_Float16*)alloc((size_t)BB * NWIN * 64 * GD * 2);
    _Float16* vlwT   = (_Float16*)alloc((size_t)BB * NWIN * GD * 64 * 2);
    _Float16* gout   = (_Float16*)alloc((size_t)BN * GD * 2);
    _Float16* lout   = (_Float16*)alloc((size_t)BN * GD * 2);
    float*    x1     = (float*)   alloc((size_t)BN * CDIM * 4);
    _Float16* hbuf   = (_Float16*)alloc((size_t)BN * HID * 2);
    _Float16* hconv  = (_Float16*)alloc((size_t)BN * HID * 2);
    _Float16* GqW16  = (_Float16*)alloc(CDIM * GD * 2);
    _Float16* GkvW16 = (_Float16*)alloc(CDIM * 2 * GD * 2);
    _Float16* GpW16  = (_Float16*)alloc(GD * GD * 2);
    _Float16* LqW16  = (_Float16*)alloc(CDIM * GD * 2);
    _Float16* LkvW16 = (_Float16*)alloc(CDIM * 2 * GD * 2);
    _Float16* LpW16  = (_Float16*)alloc(GD * GD * 2);
    _Float16* f1W16  = (_Float16*)alloc(CDIM * HID * 2);
    _Float16* f2W16  = (_Float16*)alloc(HID * CDIM * 2);
    _Float16* h2in   = xn;   // xn dead after Lq

    auto cvt = [&](const float* s, _Float16* d, int K, int N) {
        cvt_swz_kernel<<<(K * N + 255) / 256, 256, 0, stream>>>(s, d, K, N);
    };
    cvt(GqW, GqW16, CDIM, GD);       cvt(GkvW, GkvW16, CDIM, 2 * GD);
    cvt(GpW, GpW16, GD, GD);         cvt(LqW, LqW16, CDIM, GD);
    cvt(LkvW, LkvW16, CDIM, 2 * GD); cvt(LpW, LpW16, GD, GD);
    cvt(f1W, f1W16, CDIM, HID);      cvt(f2W, f2W16, HID, CDIM);

    ln_f16_kernel<<<BN / 8, 256, 0, stream>>>(x,    g1, b1, xn);
    ln_f16_kernel<<<BN / 8, 256, 0, stream>>>(xrev, g1, b1, xrn);
    pool_kernel<<<BB * NPOOL, 256, 0, stream>>>(xrn, pooled);

    auto gemm = [&](const _Float16* A, const _Float16* W, const float* bias,
                    _Float16* OutH, float* OutF, const float* Res,
                    int M, int N, int K, int colOff) {
        dim3 grid(M / 256, N / 64);
        gemm_wmma_kernel<<<grid, 256, 0, stream>>>(A, W, bias, OutH, OutF, Res, N, K, colOff);
    };

    gemm(xn,     GqW16,  Gqb,  qb,   nullptr, nullptr, BN,         GD,     CDIM, 0);
    gemm(pooled, GkvW16, Gkvb, kvb,  nullptr, nullptr, BB * NPOOL, 2 * GD, CDIM, 0);
    gemm(xn,     LqW16,  Lqb,  qlb,  nullptr, nullptr, BN,         GD,     CDIM, 0);
    gemm(xrn,    LkvW16, Lkvb, kvlb, nullptr, nullptr, BN,         2 * GD, CDIM, 0);

    vtrans_kernel<<<(BB * NHEAD * HD * 256) / 256, 256, 0, stream>>>(kvb, vT);
    lrepack_kernel<<<BB * NWIN, 256, 0, stream>>>(qlb, kvlb, qlw, klw, vlwT);

    gattn_kernel<<<dim3(NTOK / 16 / 4, NHEAD, BB), 128, 0, stream>>>(qb, kvb, vT, gout);
    lattn_kernel<<<dim3(NWIN / 4, NHEAD, BB), 128, 0, stream>>>(qlw, klw, vlwT, lout);

    gemm(gout, GpW16, Gpb, nullptr, x1, x, BN, GD, GD, 0);
    gemm(lout, LpW16, Lpb, nullptr, x1, x, BN, GD, GD, GD);

    ln_f16_kernel<<<BN / 8, 256, 0, stream>>>(x1, g2, b2, h2in);
    gemm(h2in, f1W16, f1b, hbuf, nullptr, nullptr, BN, HID, CDIM, 0);
    dwconv_gelu_kernel<<<dim3(IMG, IMG, BB), 256, 0, stream>>>(hbuf, dwW, dwb, hconv);
    gemm(hconv, f2W16, f2b, nullptr, out, x1, BN, CDIM, HID, 0);
}
